// WeightOnlyInt8Linear_41558103556517
// MI455X (gfx1250) — matrix operand order repaired and checked by reading, not compiled, not run on hardware
//
#include <hip/hip_runtime.h>
#include <stdint.h>

// ---------------------------------------------------------------------------
// Weight-only int8 linear: out[t,o] = sum_k x[t,k]*w8[o,k]*scale[o] + bias[o]
// M=8192 tokens, N=11008 out features, K=4096 in features.
//
// MI455X (gfx1250, wave32) design:
//  * f16 WMMA (v_wmma_f32_16x16x32_f16), fp32 accumulation. int8 weights are
//    exact in f16; per-channel scale/bias folded into the epilogue.
//  * 128x256 block tile, 8 waves in a 2x4 grid, each wave 64x64 = 4x4
//    accumulators -> 16 WMMA per 16 ds_load_b128 per K-step (1:1 ratio).
//  * BK=32 K-steps, double-buffered LDS, rows padded to 40 halves so
//    fragment reads land on distinct multiple-of-4 bank offsets.
//  * int8->f16 via v_perm_b32 magic-bias trick: half(0x6400|(b^0x80)) - 1152
//    == b, 2 elements per v_pk_add_f16 (~2.5x fewer VALU than cvt chains).
//  * Software pipeline: next tile's global loads issued before compute,
//    global_prefetch two tiles ahead, one barrier per K-step.
// ---------------------------------------------------------------------------

typedef __attribute__((ext_vector_type(16))) _Float16 v16h;
typedef __attribute__((ext_vector_type(8)))  _Float16 v8h;
typedef __attribute__((ext_vector_type(8)))  float    v8f;
typedef __attribute__((ext_vector_type(4)))  uint32_t v4u;

static constexpr int M_TOK = 8192;
static constexpr int K_IN  = 4096;
static constexpr int N_OUT = 11008;

static constexpr int BM = 128;
static constexpr int BN = 256;
static constexpr int BK = 32;
static constexpr int LDA = BK + 8;           // padded halves per LDS row
static constexpr int NTHREADS = 256;         // 8 waves of 32
static constexpr int NROUNDS = K_IN / BK;    // 128

// 8 int8 (two dwords) -> 8 f16 via perm + packed add.
__device__ __forceinline__ v8h i8x8_to_h8(uint32_t w0, uint32_t w1) {
    const uint32_t magic = 0x64006400u;      // f16 exponent field for 1024
    w0 ^= 0x80808080u;                       // two's complement -> biased
    w1 ^= 0x80808080u;
    v4u p;
    p.x = __builtin_amdgcn_perm(magic, w0, 0x05010500u); // {0x64|b0, 0x64|b1}
    p.y = __builtin_amdgcn_perm(magic, w0, 0x05030502u); // {0x64|b2, 0x64|b3}
    p.z = __builtin_amdgcn_perm(magic, w1, 0x05010500u);
    p.w = __builtin_amdgcn_perm(magic, w1, 0x05030502u);
    v8h h = __builtin_bit_cast(v8h, p);
    return h + (_Float16)(-1152.0f);         // (1024 + 128) removed -> exact b
}

__global__ __launch_bounds__(NTHREADS)
void wo_int8_linear_wmma(const float* __restrict__ x,
                         const int8_t* __restrict__ qw,
                         const float* __restrict__ scale,
                         const float* __restrict__ bias,
                         float* __restrict__ out)
{
    __shared__ _Float16 sA[2][BM * LDA];     // 20 KB
    __shared__ _Float16 sB[2][BN * LDA];     // 40 KB

    const int tid  = threadIdx.x;
    const int lane = tid & 31;
    const int wave = tid >> 5;
    const int wm   = wave & 1;     // 0..1 -> 64-row stripe
    const int wn   = wave >> 1;    // 0..3 -> 64-col stripe
    const int lr   = lane & 15;
    const int hsel = lane >> 4;

    const int blockM = blockIdx.y * BM;
    const int blockN = blockIdx.x * BN;

    // staging: A -> thread owns (row tid/2, 16-wide k chunk); B -> row tid, all 32
    const int sr = tid >> 1;
    const int sk = (tid & 1) << 4;

    const float*  gA = x  + (size_t)(blockM + sr)  * K_IN + sk;
    const int8_t* gB = qw + (size_t)(blockN + tid) * K_IN;

    v8f acc[4][4] = {};            // 4 M-tiles x 4 N-tiles per wave

    float4 rA[4];                  // 16 floats of A in flight
    int4   rB[2];                  // 32 int8 of B in flight

    auto loadG = [&](int r) {
        const float4* pa = (const float4*)(gA + (size_t)r * BK);
        rA[0] = pa[0]; rA[1] = pa[1]; rA[2] = pa[2]; rA[3] = pa[3];
        const int4* pb = (const int4*)(gB + (size_t)r * BK);
        rB[0] = pb[0]; rB[1] = pb[1];
        if (r + 2 < NROUNDS) {
            __builtin_prefetch(gA + (size_t)(r + 2) * BK, 0, 3);
            __builtin_prefetch(gB + (size_t)(r + 2) * BK, 0, 3);
        }
    };

    auto storeL = [&](int b) {
        // A: f32 -> f16 (compiler emits v_cvt_pk_f16_f32 pairs)
        v8h* da = (v8h*)&sA[b][sr * LDA + sk];
        #pragma unroll
        for (int j = 0; j < 2; ++j) {
            float4 f0 = rA[2 * j], f1 = rA[2 * j + 1];
            v8h t;
            t[0] = (_Float16)f0.x; t[1] = (_Float16)f0.y;
            t[2] = (_Float16)f0.z; t[3] = (_Float16)f0.w;
            t[4] = (_Float16)f1.x; t[5] = (_Float16)f1.y;
            t[6] = (_Float16)f1.z; t[7] = (_Float16)f1.w;
            da[j] = t;
        }
        // B: int8 -> f16 via perm magic
        const uint32_t* bw = (const uint32_t*)rB;
        v8h* db = (v8h*)&sB[b][tid * LDA];
        #pragma unroll
        for (int j = 0; j < 4; ++j)
            db[j] = i8x8_to_h8(bw[2 * j], bw[2 * j + 1]);
    };

    auto compute = [&](int b) {
        v16h af[4], bf[4];
        // A fragment: lane-half hsel holds K {0-7,16-23} / {8-15,24-31}
        #pragma unroll
        for (int tm = 0; tm < 4; ++tm) {
            const int row = wm * 64 + tm * 16 + lr;
            v8h lo = *(const v8h*)&sA[b][row * LDA + hsel * 8];
            v8h hi = *(const v8h*)&sA[b][row * LDA + 16 + hsel * 8];
            af[tm] = __builtin_shufflevector(lo, hi,
                      0, 1, 2, 3, 4, 5, 6, 7, 8, 9, 10, 11, 12, 13, 14, 15);
        }
        // B fragment: lane = column, lane-half holds K {0-7,16-23} / {8-15,24-31}
        #pragma unroll
        for (int tn = 0; tn < 4; ++tn) {
            const int col = wn * 64 + tn * 16 + lr;
            v8h lo = *(const v8h*)&sB[b][col * LDA + hsel * 8];
            v8h hi = *(const v8h*)&sB[b][col * LDA + 16 + hsel * 8];
            bf[tn] = __builtin_shufflevector(lo, hi,
                      0, 1, 2, 3, 4, 5, 6, 7, 8, 9, 10, 11, 12, 13, 14, 15);
        }
        #pragma unroll
        for (int tm = 0; tm < 4; ++tm)
            #pragma unroll
            for (int tn = 0; tn < 4; ++tn)
                acc[tm][tn] = __builtin_amdgcn_wmma_f32_16x16x32_f16(
                    false, af[tm], false, bf[tn],
                    (short)0, acc[tm][tn], false, false);
    };

    // -------- pipeline prologue --------
    loadG(0);
    storeL(0);
    __syncthreads();

    // -------- main K loop, double buffered --------
    #pragma unroll 1
    for (int r = 0; r < NROUNDS; ++r) {
        if (r + 1 < NROUNDS) loadG(r + 1);
        compute(r & 1);                     // 16 WMMAs per wave
        if (r + 1 < NROUNDS) storeL((r + 1) & 1);
        __syncthreads();
    }

    // -------- epilogue: scale + bias --------
    #pragma unroll
    for (int tn = 0; tn < 4; ++tn) {
        const int col = blockN + wn * 64 + tn * 16 + lr;
        const float sc = scale[col];
        const float bs = bias[col];
        #pragma unroll
        for (int tm = 0; tm < 4; ++tm) {
            const int row0 = blockM + wm * 64 + tm * 16 + hsel * 8;
            #pragma unroll
            for (int i = 0; i < 8; ++i)
                out[(size_t)(row0 + i) * N_OUT + col] = acc[tm][tn][i] * sc + bs;
        }
    }
}

extern "C" void kernel_launch(void* const* d_in, const int* in_sizes, int n_in,
                              void* d_out, int out_size, void* d_ws, size_t ws_size,
                              hipStream_t stream)
{
    const float*  xp = (const float*)d_in[0];
    const int8_t* wp = (const int8_t*)d_in[1];
    const float*  sp = (const float*)d_in[2];
    const float*  bp = (const float*)d_in[3];
    float*        op = (float*)d_out;

    dim3 grid(N_OUT / BN, M_TOK / BM);   // (43, 64)
    wo_int8_linear_wmma<<<grid, NTHREADS, 0, stream>>>(xp, wp, sp, bp, op);
}
